// Graph_Net_40888088658218
// MI455X (gfx1250) — matrix operand order repaired and checked by reading, not compiled, not run on hardware
//
#include <hip/hip_runtime.h>

typedef __attribute__((ext_vector_type(16))) _Float16 v16h;
typedef __attribute__((ext_vector_type(8)))  _Float16 v8h;
typedef __attribute__((ext_vector_type(4)))  _Float16 v4h;
typedef __attribute__((ext_vector_type(8)))  float    v8f;
typedef __attribute__((ext_vector_type(4)))  float    v4f;

#define B_  32
#define N1  1024
#define KP1 256
#define KP2 64

enum { F_RELU = 1, F_BIAS = 2, F_ROWSCALE = 4, F_SQSUM = 8, F_TRANSA = 16, F_STORE = 32 };

// ---------------------------------------------------------------------------
// Generic batched GEMM, D = epilogue(A @ B), f16 WMMA with f32 accumulate.
// Tile: 64x64x32, 128 threads = 4 waves, each wave owns a 32x32 quadrant
// (2x2 v_wmma_f32_16x16x32_f16). FLAGS is a compile-time template parameter
// so each instantiation carries only its own staging/epilogue code.
// ---------------------------------------------------------------------------
template <int FLAGS>
__global__ __launch_bounds__(128)
void k_gemm(const float* __restrict__ A, long long bsA, int ldA,
            const float* __restrict__ Bm, long long bsB, int ldB,
            const float* __restrict__ bias,
            const float* __restrict__ rowscale, long long bsS,
            float* __restrict__ D, long long bsD, int ldD,
            int M, int N, int K, float* __restrict__ sqacc)
{
    // row stride 40 halves = 80 B: each lane's 8-half runs stay 16B-aligned
    __shared__ _Float16 As[64][40];    // As[m][k]
    __shared__ _Float16 BsT[64][40];   // BsT[n][k]  (B stored transposed)
    __shared__ float    red[4];

    const int b  = blockIdx.z;
    const int m0 = blockIdx.y * 64;
    const int n0 = blockIdx.x * 64;
    const float* Ab = A  + (long long)b * bsA;
    const float* Bb = Bm + (long long)b * bsB;

    const int t    = threadIdx.x;
    const int lane = t & 31;
    const int wave = t >> 5;
    const int wr   = (wave >> 1) * 32;
    const int wc   = (wave & 1)  * 32;
    const int lr   = lane & 15;
    const int hi   = lane >> 4;

    v8f zero = {};
    v8f acc[2][2];
    acc[0][0] = zero; acc[0][1] = zero; acc[1][0] = zero; acc[1][1] = zero;

    for (int k0 = 0; k0 < K; k0 += 32) {
        // ---- stage A tile (64 x 32) ----
        if (FLAGS & F_TRANSA) {
            // A[m][k] = Aglob[k][m]; consecutive threads sweep m (coalesced)
            for (int i = t; i < 64 * 32; i += 128) {
                int kk = i >> 6, mm = i & 63;
                int mg = m0 + mm, kg = k0 + kk;
                float v = (mg < M && kg < K) ? Ab[(long long)kg * ldA + mg] : 0.0f;
                As[mm][kk] = (_Float16)v;
            }
        } else {
            // vectorized: one v4f per thread-iteration (K, ld multiples of 4)
            for (int i = t; i < 64 * 8; i += 128) {
                int mm = i >> 3, kq = (i & 7) * 4;
                int mg = m0 + mm, kg = k0 + kq;
                v4f v = {};
                if (mg < M) {
                    const float* p = Ab + (long long)mg * ldA + kg;
                    if (kg + 3 < K) v = *(const v4f*)p;
                    else {
                        if (kg + 0 < K) v.x = p[0];
                        if (kg + 1 < K) v.y = p[1];
                        if (kg + 2 < K) v.z = p[2];
                        if (kg + 3 < K) v.w = p[3];
                    }
                }
                v4h h; h.x = (_Float16)v.x; h.y = (_Float16)v.y;
                       h.z = (_Float16)v.z; h.w = (_Float16)v.w;
                *(v4h*)(&As[mm][kq]) = h;
            }
        }
        // ---- stage B tile (32 x 64) transposed into BsT[n][k] ----
        for (int i = t; i < 32 * 16; i += 128) {
            int kk = i >> 4, nq = (i & 15) * 4;
            int kg = k0 + kk, ng = n0 + nq;
            v4f v = {};
            if (kg < K) {
                const float* p = Bb + (long long)kg * ldB + ng;
                if (ng + 3 < N) v = *(const v4f*)p;
                else {
                    if (ng + 0 < N) v.x = p[0];
                    if (ng + 1 < N) v.y = p[1];
                    if (ng + 2 < N) v.z = p[2];
                    if (ng + 3 < N) v.w = p[3];
                }
            }
            BsT[nq + 0][kk] = (_Float16)v.x;
            BsT[nq + 1][kk] = (_Float16)v.y;
            BsT[nq + 2][kk] = (_Float16)v.z;
            BsT[nq + 3][kk] = (_Float16)v.w;
        }
        __syncthreads();

        // ---- fragments via 16B LDS vector loads ----
        // A 16x32 layout: lane lr = M; hi half: elems0-7 K=hi*8+0..7, elems8-15 K=16+hi*8+0..7
        v16h af[2], bf[2];
        #pragma unroll
        for (int ia = 0; ia < 2; ia++) {
            int mb = wr + ia * 16 + lr;
            v8h lo = *(const v8h*)(&As[mb][hi * 8]);
            v8h hh = *(const v8h*)(&As[mb][16 + hi * 8]);
            af[ia] = __builtin_shufflevector(lo, hh, 0,1,2,3,4,5,6,7,8,9,10,11,12,13,14,15);
        }
        // B 32x16 layout: lane lr = N; hi half: elems0-7 K=hi*8+0..7, elems8-15 K=16+hi*8+0..7
        #pragma unroll
        for (int ib = 0; ib < 2; ib++) {
            int nb = wc + ib * 16 + lr;
            v8h lo = *(const v8h*)(&BsT[nb][hi * 8]);
            v8h hh = *(const v8h*)(&BsT[nb][16 + hi * 8]);
            bf[ib] = __builtin_shufflevector(lo, hh, 0,1,2,3,4,5,6,7,8,9,10,11,12,13,14,15);
        }
        #pragma unroll
        for (int ia = 0; ia < 2; ia++)
            #pragma unroll
            for (int ib = 0; ib < 2; ib++)
                acc[ia][ib] = __builtin_amdgcn_wmma_f32_16x16x32_f16(
                    false, af[ia], false, bf[ib], (short)0, acc[ia][ib], false, false);
        __syncthreads();
    }

    // ---- epilogue: C/D layout — lane lr is N, VGPR r is M = r + hi*8 ----
    float lsq = 0.0f;
    #pragma unroll
    for (int ia = 0; ia < 2; ia++) {
        #pragma unroll
        for (int ib = 0; ib < 2; ib++) {
            int n = n0 + wc + ib * 16 + lr;
            #pragma unroll
            for (int r = 0; r < 8; r++) {
                int m = m0 + wr + ia * 16 + hi * 8 + r;
                bool valid = (m < M) && (n < N);
                float v = acc[ia][ib][r];
                if (FLAGS & F_ROWSCALE) {
                    float dg = valid ? rowscale[(long long)b * bsS + m] : 1.0f;
                    v /= dg;
                }
                if (FLAGS & F_BIAS) v += valid ? bias[n] : 0.0f;
                if (FLAGS & F_RELU) v = v > 0.0f ? v : 0.0f;
                if ((FLAGS & F_SQSUM) && valid) lsq += v * v;
                if ((FLAGS & F_STORE) && valid)
                    D[(long long)b * bsD + (long long)m * ldD + n] = v;
            }
        }
    }
    if (FLAGS & F_SQSUM) {
        for (int off = 16; off; off >>= 1) lsq += __shfl_down(lsq, off, 32);
        if (lane == 0) red[wave] = lsq;
        __syncthreads();
        if (t == 0) atomicAdd(sqacc, red[0] + red[1] + red[2] + red[3]);
    }
}

// adj_sl = adj with diag forced to 1; deg = max(1, rowsum(adj_sl));
// also accumulate sum of original adj (== ||adj||_F^2 since adj is 0/1)
__global__ __launch_bounds__(256)
void k_adj_selfloop(const float* __restrict__ adj, float* __restrict__ adjsl,
                    float* __restrict__ deg, float* __restrict__ sumacc,
                    int n, long long rows)
{
    int wave = threadIdx.x >> 5, lane = threadIdx.x & 31;
    long long row = (long long)blockIdx.x * 8 + wave;
    if (row >= rows) return;
    int i = (int)(row % n);
    const float* src = adj   + row * (long long)n;
    float*       dst = adjsl + row * (long long)n;
    float ssl = 0.0f, sor = 0.0f;
    for (int c = lane; c < n; c += 32) {
        float v = src[c];
        sor += v;
        if (c == i) v = 1.0f;
        dst[c] = v;
        ssl += v;
    }
    for (int off = 16; off; off >>= 1) { ssl += __shfl_xor(ssl, off, 32); sor += __shfl_xor(sor, off, 32); }
    if (lane == 0) { deg[row] = fmaxf(ssl, 1.0f); atomicAdd(sumacc, sor); }
}

__global__ __launch_bounds__(256)
void k_rowsum_clip(const float* __restrict__ a, float* __restrict__ deg,
                   int n, long long rows)
{
    int wave = threadIdx.x >> 5, lane = threadIdx.x & 31;
    long long row = (long long)blockIdx.x * 8 + wave;
    if (row >= rows) return;
    const float* src = a + row * (long long)n;
    float s = 0.0f;
    for (int c = lane; c < n; c += 32) s += src[c];
    for (int off = 16; off; off >>= 1) s += __shfl_xor(s, off, 32);
    if (lane == 0) deg[row] = fmaxf(s, 1.0f);
}

// row softmax in place + accumulate entropy sum(-s*log(s+1e-15)); K in {64,256}
__global__ __launch_bounds__(256)
void k_softmax_ent(float* __restrict__ S, long long rows, int K, float* __restrict__ entacc)
{
    int wave = threadIdx.x >> 5, lane = threadIdx.x & 31;
    long long row = (long long)blockIdx.x * 8 + wave;
    if (row >= rows) return;
    float* p = S + row * (long long)K;
    int nk = K >> 5;
    float v[8];
    float mx = -1e30f;
    for (int i = 0; i < nk; i++) { v[i] = p[i * 32 + lane]; mx = fmaxf(mx, v[i]); }
    for (int off = 16; off; off >>= 1) mx = fmaxf(mx, __shfl_xor(mx, off, 32));
    float sum = 0.0f;
    for (int i = 0; i < nk; i++) { v[i] = __expf(v[i] - mx); sum += v[i]; }
    for (int off = 16; off; off >>= 1) sum += __shfl_xor(sum, off, 32);
    float inv = 1.0f / sum;
    float ent = 0.0f;
    for (int i = 0; i < nk; i++) {
        float s = v[i] * inv;
        p[i * 32 + lane] = s;
        ent -= s * __logf(s + 1e-15f);
    }
    for (int off = 16; off; off >>= 1) ent += __shfl_xor(ent, off, 32);
    if (lane == 0) atomicAdd(entacc, ent);
}

__global__ __launch_bounds__(256)
void k_dot_reduce(const float* __restrict__ a, const float* __restrict__ b,
                  long long n, float* __restrict__ acc)
{
    __shared__ float red[8];
    long long i = (long long)blockIdx.x * blockDim.x + threadIdx.x;
    long long stride = (long long)gridDim.x * blockDim.x;
    float s = 0.0f;
    for (; i < n; i += stride) s += a[i] * b[i];
    for (int off = 16; off; off >>= 1) s += __shfl_xor(s, off, 32);
    int wave = threadIdx.x >> 5, lane = threadIdx.x & 31;
    if (lane == 0) red[wave] = s;
    __syncthreads();
    if (threadIdx.x == 0) {
        float t = 0.0f;
        for (int w = 0; w < 8; w++) t += red[w];
        atomicAdd(acc, t);
    }
}

// mean over nodes; writes aligned ws copy (GEMM operand) AND the output tail
__global__ void k_mean_nodes(const float* __restrict__ h, float* __restrict__ outws,
                             float* __restrict__ outtail, int Bv, int Nn, int Cc)
{
    int idx = blockIdx.x * blockDim.x + threadIdx.x;
    if (idx >= Bv * Cc) return;
    int b = idx / Cc, c = idx % Cc;
    const float* p = h + (long long)b * Nn * Cc + c;
    float s = 0.0f;
    for (int n = 0; n < Nn; n++) s += p[(long long)n * Cc];
    float m = s / (float)Nn;
    outws[idx]   = m;
    outtail[idx] = m;
}

__global__ void k_zero(float* p, int n)
{
    int i = blockIdx.x * blockDim.x + threadIdx.x;
    if (i < n) p[i] = 0.0f;
}

// sc: 0=sum(adj1) 1=sum(s1*sa1) 2=||s1^T s1||^2 3=ent1
//     4=||adj2||^2 5=sum(s2*sa2) 6=||s2^T s2||^2 7=ent2
__global__ void k_finalize(const float* __restrict__ sc, float* __restrict__ out)
{
    if (threadIdx.x == 0 && blockIdx.x == 0) {
        float l1 = sqrtf(fmaxf(sc[0] - 2.0f * sc[1] + sc[2], 0.0f)) /
                   ((float)B_ * (float)N1 * (float)N1);
        float l2 = sqrtf(fmaxf(sc[4] - 2.0f * sc[5] + sc[6], 0.0f)) /
                   ((float)B_ * (float)KP1 * (float)KP1);
        out[1024] = l1 + l2;
        out[1025] = sc[3] / ((float)B_ * (float)N1) + sc[7] / ((float)B_ * (float)KP1);
    }
}

// ---------------------------------------------------------------------------
// Workspace layout (floats)
// ---------------------------------------------------------------------------
constexpr long long OFF_SC    = 0;
constexpr long long OFF_DEG1  = 64;
constexpr long long OFF_DEG2  = OFF_DEG1  + (long long)B_ * N1;
constexpr long long OFF_DEG3  = OFF_DEG2  + (long long)B_ * KP1;
constexpr long long OFF_GEMB  = OFF_DEG3  + (long long)B_ * KP2;
constexpr long long OFF_ADJSL = OFF_GEMB  + (long long)B_ * 192;
constexpr long long OFF_AGG   = OFF_ADJSL + (long long)B_ * N1 * N1;
constexpr long long OFF_XP1   = OFF_AGG   + (long long)B_ * N1 * 64;
constexpr long long OFF_XE1   = OFF_XP1   + (long long)B_ * N1 * 384;
constexpr long long OFF_S1    = OFF_XE1   + (long long)B_ * N1 * 192;
constexpr long long OFF_SA1   = OFF_S1    + (long long)B_ * N1 * 256;
constexpr long long OFF_H2    = OFF_SA1   + (long long)B_ * N1 * 256;
constexpr long long OFF_ADJ2  = OFF_H2    + (long long)B_ * KP1 * 192;
constexpr long long OFF_XP2   = OFF_ADJ2  + (long long)B_ * KP1 * KP1;
constexpr long long OFF_XE2   = OFF_XP2   + (long long)B_ * KP1 * 192;
constexpr long long OFF_S2    = OFF_XE2   + (long long)B_ * KP1 * 192;
constexpr long long OFF_SA2   = OFF_S2    + (long long)B_ * KP1 * KP2;
constexpr long long OFF_H3    = OFF_SA2   + (long long)B_ * KP1 * KP2;
constexpr long long OFF_ADJ3  = OFF_H3    + (long long)B_ * KP2 * 192;
constexpr long long OFF_XE3   = OFF_ADJ3  + (long long)B_ * KP2 * KP2;
constexpr long long OFF_LIN1  = OFF_XE3   + (long long)B_ * KP2 * 192;

static inline void gemm(hipStream_t st,
                        const float* A, long long bsA, int ldA,
                        const float* B, long long bsB, int ldB,
                        const float* bias,
                        const float* rs, long long bsS,
                        float* D, long long bsD, int ldD,
                        int M, int N, int K, int batch, int flags, float* sq = nullptr)
{
    dim3 g((N + 63) / 64, (M + 63) / 64, batch);
#define LAUNCH_GEMM(F) k_gemm<F><<<g, dim3(128), 0, st>>>(A, bsA, ldA, B, bsB, ldB, \
        bias, rs, bsS, D, bsD, ldD, M, N, K, sq)
    switch (flags) {
        case F_ROWSCALE | F_STORE:          LAUNCH_GEMM(F_ROWSCALE | F_STORE); break;
        case F_BIAS | F_RELU | F_STORE:     LAUNCH_GEMM(F_BIAS | F_RELU | F_STORE); break;
        case F_TRANSA | F_STORE:            LAUNCH_GEMM(F_TRANSA | F_STORE); break;
        case F_STORE:                       LAUNCH_GEMM(F_STORE); break;
        case F_TRANSA | F_SQSUM:            LAUNCH_GEMM(F_TRANSA | F_SQSUM); break;
        case F_BIAS | F_STORE:              LAUNCH_GEMM(F_BIAS | F_STORE); break;
        default: break;
    }
#undef LAUNCH_GEMM
}

// One DiffPool GNN block: 3 SAGE convs (+ optional final linear)
static void run_gnn(hipStream_t st, const float* adjp, long long bsAdj,
                    const float* deg, int nodes,
                    const float* xin, int ldX, long long bsX, int cin,
                    float* agg, float* xcat, long long bsCat, int catw, int cout3,
                    const float* const* W, bool lin, float* sout, long long bsS)
{
    // conv1
    gemm(st, adjp, bsAdj, nodes, xin, bsX, ldX, nullptr, deg, nodes,
         agg, (long long)nodes * cin, cin, nodes, cin, nodes, B_, F_ROWSCALE | F_STORE);
    gemm(st, agg, (long long)nodes * cin, cin, W[0], 0, 64, W[1], nullptr, 0,
         xcat + 0, bsCat, catw, nodes, 64, cin, B_, F_BIAS | F_RELU | F_STORE);
    // conv2
    gemm(st, adjp, bsAdj, nodes, xcat + 0, bsCat, catw, nullptr, deg, nodes,
         agg, (long long)nodes * 64, 64, nodes, 64, nodes, B_, F_ROWSCALE | F_STORE);
    gemm(st, agg, (long long)nodes * 64, 64, W[2], 0, 64, W[3], nullptr, 0,
         xcat + 64, bsCat, catw, nodes, 64, 64, B_, F_BIAS | F_RELU | F_STORE);
    // conv3
    gemm(st, adjp, bsAdj, nodes, xcat + 64, bsCat, catw, nullptr, deg, nodes,
         agg, (long long)nodes * 64, 64, nodes, 64, nodes, B_, F_ROWSCALE | F_STORE);
    gemm(st, agg, (long long)nodes * 64, 64, W[4], 0, cout3, W[5], nullptr, 0,
         xcat + 128, bsCat, catw, nodes, cout3, 64, B_, F_BIAS | F_RELU | F_STORE);
    if (lin)
        gemm(st, xcat, bsCat, catw, W[6], 0, cout3, W[7], nullptr, 0,
             sout, bsS, cout3, nodes, cout3, catw, B_, F_BIAS | F_RELU | F_STORE);
}

extern "C" void kernel_launch(void* const* d_in, const int* in_sizes, int n_in,
                              void* d_out, int out_size, void* d_ws, size_t ws_size,
                              hipStream_t stream)
{
    const float* x   = (const float*)d_in[0];
    const float* adj = (const float*)d_in[1];
    const float* P[40];
    for (int i = 2; i < 40 && i < n_in; i++) P[i] = (const float*)d_in[i];

    float* ws  = (float*)d_ws;
    float* out = (float*)d_out;

    float* sc    = ws + OFF_SC;
    float* deg1  = ws + OFF_DEG1;
    float* deg2  = ws + OFF_DEG2;
    float* deg3  = ws + OFF_DEG3;
    float* gemb  = ws + OFF_GEMB;
    float* adjsl = ws + OFF_ADJSL;
    float* agg   = ws + OFF_AGG;
    float* xp1   = ws + OFF_XP1;
    float* xe1   = ws + OFF_XE1;
    float* s1    = ws + OFF_S1;
    float* sa1   = ws + OFF_SA1;
    float* h2    = ws + OFF_H2;
    float* adj2  = ws + OFF_ADJ2;
    float* xp2   = ws + OFF_XP2;
    float* xe2   = ws + OFF_XE2;
    float* s2    = ws + OFF_S2;
    float* sa2   = ws + OFF_SA2;
    float* h3    = ws + OFF_H3;
    float* adj3  = ws + OFF_ADJ3;
    float* xe3   = ws + OFF_XE3;
    float* lin1b = ws + OFF_LIN1;

    // 0) zero scalar accumulators (deterministic across graph replays)
    k_zero<<<1, 64, 0, stream>>>(sc, 64);

    // 1) self-loop adjacency + degree + sum(adj)
    k_adj_selfloop<<<(B_ * N1 + 7) / 8, 256, 0, stream>>>(adj, adjsl, deg1, &sc[0],
                                                          N1, (long long)B_ * N1);

    // 2) level-1 GNNs
    const float* W1p[8] = {P[2], P[3], P[4], P[5], P[6], P[7], P[8], P[9]};
    run_gnn(stream, adjsl, (long long)N1 * N1, deg1, N1,
            x, 64, (long long)N1 * 64, 64,
            agg, xp1, (long long)N1 * 384, 384, 256, W1p, true,
            s1, (long long)N1 * 256);
    const float* W1e[8] = {P[10], P[11], P[12], P[13], P[14], P[15], nullptr, nullptr};
    run_gnn(stream, adjsl, (long long)N1 * N1, deg1, N1,
            x, 64, (long long)N1 * 64, 64,
            agg, xe1, (long long)N1 * 192, 192, 64, W1e, false, nullptr, 0);

    // 3) softmax(s1) + entropy
    k_softmax_ent<<<(B_ * N1 + 7) / 8, 256, 0, stream>>>(s1, (long long)B_ * N1, 256, &sc[3]);

    // 4) diffpool-1
    gemm(stream, s1, (long long)N1 * 256, 256, xe1, (long long)N1 * 192, 192, nullptr, nullptr, 0,
         h2, (long long)KP1 * 192, 192, KP1, 192, N1, B_, F_TRANSA | F_STORE);          // h2 = s^T h
    gemm(stream, adj, (long long)N1 * N1, N1, s1, (long long)N1 * 256, 256, nullptr, nullptr, 0,
         sa1, (long long)N1 * 256, 256, N1, 256, N1, B_, F_STORE);                      // sa = adj s
    gemm(stream, s1, (long long)N1 * 256, 256, sa1, (long long)N1 * 256, 256, nullptr, nullptr, 0,
         adj2, (long long)KP1 * KP1, KP1, KP1, KP1, N1, B_, F_TRANSA | F_STORE);        // adj2 = s^T sa
    k_dot_reduce<<<1024, 256, 0, stream>>>(s1, sa1, (long long)B_ * N1 * 256, &sc[1]);  // sum(s*sa)
    gemm(stream, s1, (long long)N1 * 256, 256, s1, (long long)N1 * 256, 256, nullptr, nullptr, 0,
         nullptr, 0, KP1, KP1, KP1, N1, B_, F_TRANSA | F_SQSUM, &sc[2]);                // ||s^T s||^2

    // 5) degree of adj2
    k_rowsum_clip<<<(B_ * KP1 + 7) / 8, 256, 0, stream>>>(adj2, deg2, KP1, (long long)B_ * KP1);

    // 6) level-2 GNNs
    const float* W2p[8] = {P[16], P[17], P[18], P[19], P[20], P[21], P[22], P[23]};
    run_gnn(stream, adj2, (long long)KP1 * KP1, deg2, KP1,
            h2, 192, (long long)KP1 * 192, 192,
            agg, xp2, (long long)KP1 * 192, 192, 64, W2p, true,
            s2, (long long)KP1 * 64);
    const float* W2e[8] = {P[24], P[25], P[26], P[27], P[28], P[29], nullptr, nullptr};
    run_gnn(stream, adj2, (long long)KP1 * KP1, deg2, KP1,
            h2, 192, (long long)KP1 * 192, 192,
            agg, xe2, (long long)KP1 * 192, 192, 64, W2e, false, nullptr, 0);

    // 7) softmax(s2) + entropy
    k_softmax_ent<<<(B_ * KP1 + 7) / 8, 256, 0, stream>>>(s2, (long long)B_ * KP1, 64, &sc[7]);

    // 8) diffpool-2
    gemm(stream, s2, (long long)KP1 * 64, 64, xe2, (long long)KP1 * 192, 192, nullptr, nullptr, 0,
         h3, (long long)KP2 * 192, 192, KP2, 192, KP1, B_, F_TRANSA | F_STORE);
    gemm(stream, adj2, (long long)KP1 * KP1, KP1, s2, (long long)KP1 * 64, 64, nullptr, nullptr, 0,
         sa2, (long long)KP1 * 64, 64, KP1, 64, KP1, B_, F_STORE);
    gemm(stream, s2, (long long)KP1 * 64, 64, sa2, (long long)KP1 * 64, 64, nullptr, nullptr, 0,
         adj3, (long long)KP2 * KP2, KP2, KP2, KP2, KP1, B_, F_TRANSA | F_STORE);
    k_dot_reduce<<<256, 256, 0, stream>>>(s2, sa2, (long long)B_ * KP1 * 64, &sc[5]);
    gemm(stream, s2, (long long)KP1 * 64, 64, s2, (long long)KP1 * 64, 64, nullptr, nullptr, 0,
         nullptr, 0, KP2, KP2, KP2, KP1, B_, F_TRANSA | F_SQSUM, &sc[6]);
    k_dot_reduce<<<1024, 256, 0, stream>>>(adj2, adj2, (long long)B_ * KP1 * KP1, &sc[4]); // ||adj2||^2

    // 9) degree of adj3
    k_rowsum_clip<<<(B_ * KP2 + 7) / 8, 256, 0, stream>>>(adj3, deg3, KP2, (long long)B_ * KP2);

    // 10) level-3 embed GNN
    const float* W3e[8] = {P[30], P[31], P[32], P[33], P[34], P[35], nullptr, nullptr};
    run_gnn(stream, adj3, (long long)KP2 * KP2, deg3, KP2,
            h3, 192, (long long)KP2 * 192, 192,
            agg, xe3, (long long)KP2 * 192, 192, 64, W3e, false, nullptr, 0);

    // 11) mean over nodes -> gnn_embed (aligned ws copy + output tail)
    k_mean_nodes<<<(B_ * 192 + 255) / 256, 256, 0, stream>>>(xe3, gemb, out + 1026, B_, KP2, 192);

    // 12) classifier head (batch dim == GEMM M)
    gemm(stream, gemb, 0, 192, P[36], 0, 64, P[37], nullptr, 0,
         lin1b, 0, 64, 32, 64, 192, 1, F_BIAS | F_RELU | F_STORE);
    gemm(stream, lin1b, 0, 64, P[38], 0, 32, P[39], nullptr, 0,
         out, 0, 32, 32, 32, 64, 1, F_BIAS | F_STORE);

    // 13) losses
    k_finalize<<<1, 1, 0, stream>>>(sc, out);
}
